// MIXFFT_55284819034278
// MI455X (gfx1250) — matrix operand order repaired and checked by reading, not compiled, not run on hardware
//
#include <hip/hip_runtime.h>
#include <hip/hip_bf16.h>

// out[n,b,c] = h[n]*t[c]*Re(FFT_N FFT_C query)[n,b,c]
//            = h[n]*t[c]*(C·Q_b·C - S·Q_b·S)[n,c]   (C/S = 512x512 cos/sin DFT mats, symmetric)
// Stage 1: U = C·Q, V = S·Q      (M=512, K=512, N=B*C=32768), 16x64 tile / wave
// Stage 2: out = diag(h)(U·C - V·S)diag(t)  (M=N*B=32768, K=512, N=512), 32x32 tile / wave

typedef __attribute__((ext_vector_type(16))) __bf16 bf16x16;
typedef __attribute__((ext_vector_type(8)))  __bf16 bf16x8;
typedef __attribute__((ext_vector_type(8)))  float  f32x8;

#define NSEQ 512
#define BATCH 64
#define CDIM 512
#define BIGN (BATCH * CDIM)   // 32768

__device__ __forceinline__ f32x8 wmma_bf16(bf16x16 a, bf16x16 b, f32x8 c) {
    // 8 args: (neg_a, A, neg_b, B, c_mod, C, reuse_a, reuse_b)
    return __builtin_amdgcn_wmma_f32_16x16x32_bf16(false, a, false, b, (short)0, c,
                                                   false, false);
}

// A-matrix 16x32 bf16 fragment (row = lane%16):
//   lanegroup 0 (lanes 0-15):  VGPR0-3 -> K=0..7,  VGPR4-7 -> K=16..23
//   lanegroup 1 (lanes 16-31): VGPR0-3 -> K=8..15, VGPR4-7 -> K=24..31
// => two contiguous 16-byte chunks per lane.
__device__ __forceinline__ bf16x16 load_a_frag(const __bf16* __restrict__ base, int lda,
                                               int row0, int k0, int lane) {
    const int row = row0 + (lane & 15);
    const int lg  = lane >> 4;
    const __bf16* p = base + (size_t)row * lda + k0 + 8 * lg;
    bf16x8 lo = *(const bf16x8*)(p);
    bf16x8 hi = *(const bf16x8*)(p + 16);
    bf16x16 f;
#pragma unroll
    for (int i = 0; i < 8; ++i) { f[i] = lo[i]; f[i + 8] = hi[i]; }
    return f;
}

// B-matrix 32x16 bf16 fragment (col = lane%16), loaded from a K-contiguous
// (i.e. transposed/symmetric) array:
//   lanegroup 0: K=0..7, 16..23; lanegroup 1: K=8..15, 24..31 (same as A)
// => two contiguous 16-byte chunks per lane.
__device__ __forceinline__ bf16x16 load_b_frag(const __bf16* __restrict__ baseT, int ldb,
                                               int col0, int k0, int lane) {
    const int col = col0 + (lane & 15);
    const int lg  = lane >> 4;
    const __bf16* p = baseT + (size_t)col * ldb + k0 + 8 * lg;
    bf16x8 lo = *(const bf16x8*)(p);
    bf16x8 hi = *(const bf16x8*)(p + 16);
    bf16x16 f;
#pragma unroll
    for (int i = 0; i < 8; ++i) { f[i] = lo[i]; f[i + 8] = hi[i]; }
    return f;
}

// ---- Kernel 1: cos/sin DFT tables in bf16 (symmetric 512x512) ----
__global__ void MIXFFT_tables(__bf16* __restrict__ ct, __bf16* __restrict__ st) {
    int idx = blockIdx.x * 256 + threadIdx.x;      // 0 .. 512*512-1
    int n = idx >> 9, m = idx & 511;
    int p = (n * m) & 511;
    float theta = (float)p * 0.0122718463030851f;  // 2*pi/512
    ct[idx] = (__bf16)cosf(theta);
    st[idx] = (__bf16)sinf(theta);
}

// ---- Kernel 2: Qt[j, m] = query[m, b, c] in bf16, j = b*512+c (LDS tile transpose) ----
__global__ void MIXFFT_transpose(const float* __restrict__ q, __bf16* __restrict__ qt) {
    __shared__ __bf16 tile[32][33];
    const int jt = blockIdx.x * 32;   // over j (0..32767)
    const int mt = blockIdx.y * 32;   // over m (0..511)
    const int tx = threadIdx.x;       // 0..31
    const int ty = threadIdx.y;       // 0..7
#pragma unroll
    for (int k = 0; k < 4; ++k) {
        int m = mt + ty + 8 * k;
        int j = jt + tx;
        tile[ty + 8 * k][tx] = (__bf16)q[(size_t)m * BIGN + j];
    }
    __syncthreads();
#pragma unroll
    for (int k = 0; k < 4; ++k) {
        int j = jt + ty + 8 * k;
        int m = mt + tx;
        qt[(size_t)j * NSEQ + m] = tile[tx][ty + 8 * k];
    }
}

// ---- Kernel 3: U = C*Q, V = S*Q. 16x64 tile per wave; A-frags reused 4x. ----
__global__ void __launch_bounds__(128)
MIXFFT_gemm1(const __bf16* __restrict__ Ctab, const __bf16* __restrict__ Stab,
             const __bf16* __restrict__ Qt, __bf16* __restrict__ U,
             __bf16* __restrict__ V) {
    const int lane = threadIdx.x & 31;
    const int wave = threadIdx.x >> 5;
    const int g    = blockIdx.x * 4 + wave;   // 32 row-tiles * 512 col-groups
    const int mt   = g >> 9;                  // 0..31  (rows n)
    const int ng   = g & 511;                 // 0..511 (col group of 64)
    const int row0 = mt * 16;
    const int col0 = ng * 64;

    f32x8 accC[4] = {};
    f32x8 accS[4] = {};
    for (int k0 = 0; k0 < NSEQ; k0 += 32) {
        __builtin_prefetch(Qt + (size_t)(col0 + (lane & 15)) * NSEQ + k0 + 128, 0, 1);
        bf16x16 aC = load_a_frag(Ctab, NSEQ, row0, k0, lane);
        bf16x16 aS = load_a_frag(Stab, NSEQ, row0, k0, lane);
#pragma unroll
        for (int t = 0; t < 4; ++t) {
            bf16x16 b = load_b_frag(Qt, NSEQ, col0 + 16 * t, k0, lane);
            accC[t] = wmma_bf16(aC, b, accC[t]);
            accS[t] = wmma_bf16(aS, b, accS[t]);
        }
    }

    const int lg = lane >> 4;
#pragma unroll
    for (int t = 0; t < 4; ++t) {
        const int cw = col0 + 16 * t + (lane & 15);
#pragma unroll
        for (int r = 0; r < 8; ++r) {
            int row = row0 + r + 8 * lg;
            size_t off = (size_t)row * BIGN + cw;
            U[off] = (__bf16)accC[t][r];
            V[off] = (__bf16)accS[t][r];
        }
    }
}

// ---- Kernel 4: out = diag(h)*(U*C - V*S)*diag(t). 32x32 tile per wave. ----
__global__ void __launch_bounds__(128)
MIXFFT_gemm2(const __bf16* __restrict__ U, const __bf16* __restrict__ V,
             const __bf16* __restrict__ Ctab, const __bf16* __restrict__ Stab,
             const float* __restrict__ h, const float* __restrict__ t,
             float* __restrict__ out) {
    const int lane = threadIdx.x & 31;
    const int wave = threadIdx.x >> 5;
    const int g    = blockIdx.x * 4 + wave;   // 1024 row-groups * 16 col-groups
    const int rg   = g >> 4;                  // 0..1023 (rows i = n*64+b, group of 32)
    const int cg   = g & 15;                  // 0..15   (cols c', group of 32)
    const int row0 = rg * 32;
    const int col0 = cg * 32;

    f32x8 accU[2][2] = {};
    f32x8 accV[2][2] = {};
    for (int k0 = 0; k0 < CDIM; k0 += 32) {
        __builtin_prefetch(U + (size_t)(row0 + (lane & 15)) * CDIM + k0 + 128, 0, 1);
        bf16x16 aU[2], aV[2], bC[2], bS[2];
#pragma unroll
        for (int i = 0; i < 2; ++i) {
            aU[i] = load_a_frag(U, CDIM, row0 + 16 * i, k0, lane);
            aV[i] = load_a_frag(V, CDIM, row0 + 16 * i, k0, lane);
            bC[i] = load_b_frag(Ctab, CDIM, col0 + 16 * i, k0, lane);  // C symmetric
            bS[i] = load_b_frag(Stab, CDIM, col0 + 16 * i, k0, lane);  // S symmetric
        }
#pragma unroll
        for (int i = 0; i < 2; ++i)
#pragma unroll
            for (int j = 0; j < 2; ++j) {
                accU[i][j] = wmma_bf16(aU[i], bC[j], accU[i][j]);
                accV[i][j] = wmma_bf16(aV[i], bS[j], accV[i][j]);
            }
    }

    const int lg = lane >> 4;
#pragma unroll
    for (int j = 0; j < 2; ++j) {
        const int col = col0 + 16 * j + (lane & 15);
        const float tc = t[col];
#pragma unroll
        for (int i = 0; i < 2; ++i)
#pragma unroll
            for (int r = 0; r < 8; ++r) {
                int row = row0 + 16 * i + r + 8 * lg;    // i' = n*64 + b
                float s = h[row >> 6] * tc;              // n = i'/64
                out[(size_t)row * CDIM + col] = (accU[i][j][r] - accV[i][j][r]) * s;
            }
    }
}

extern "C" void kernel_launch(void* const* d_in, const int* in_sizes, int n_in,
                              void* d_out, int out_size, void* d_ws, size_t ws_size,
                              hipStream_t stream) {
    const float* q = (const float*)d_in[0];   // (512, 64, 512) f32
    const float* h = (const float*)d_in[1];   // (512,) f32 (hidden_w flattened)
    const float* t = (const float*)d_in[2];   // (512,) f32 (token_w flattened)
    float* out = (float*)d_out;               // (512, 64, 512) f32

    char* ws = (char*)d_ws;
    __bf16* Ctab = (__bf16*)(ws + 0);                               // 512 KB
    __bf16* Stab = (__bf16*)(ws + (size_t)524288);                  // 512 KB
    __bf16* Qt   = (__bf16*)(ws + (size_t)1048576);                 // 32 MB
    __bf16* U    = (__bf16*)(ws + (size_t)1048576 + 33554432);      // 32 MB
    __bf16* V    = (__bf16*)(ws + (size_t)1048576 + 2 * 33554432);  // 32 MB

    MIXFFT_tables<<<1024, 256, 0, stream>>>(Ctab, Stab);
    dim3 tgrid(BIGN / 32, NSEQ / 32);
    MIXFFT_transpose<<<tgrid, dim3(32, 8), 0, stream>>>(q, Qt);
    MIXFFT_gemm1<<<(32 * 512) / 4, 128, 0, stream>>>(Ctab, Stab, Qt, U, V);
    MIXFFT_gemm2<<<(1024 * 16) / 4, 128, 0, stream>>>(U, V, Ctab, Stab, h, t, out);
}
